// MultiHeadedCrossAttention_4509715660805
// MI455X (gfx1250) — matrix operand order repaired and checked by reading, not compiled, not run on hardware
//
#include <hip/hip_runtime.h>
#include <stdint.h>

// ---------- types ----------
typedef __attribute__((ext_vector_type(16))) __bf16    bf16x16;
typedef __attribute__((ext_vector_type(8)))  float     floatx8;
typedef __attribute__((ext_vector_type(4)))  uint32_t  u32x4;
typedef __attribute__((ext_vector_type(4)))  float     f32x4;
typedef __attribute__((ext_vector_type(4)))  uint16_t  u16x4;

union BfVec {           // 32 bytes: one WMMA 16-bit operand per lane
  u32x4   q[2];
  bf16x16 v;
};
union Pack8 {           // 8 bf16 packed for one 16B store
  uint16_t s[8];
  u32x4    q;
};

__device__ __forceinline__ uint16_t f2bf(float f) {
  uint32_t u = __float_as_uint(f);
  u += 0x7FFFu + ((u >> 16) & 1u);          // round-to-nearest-even
  return (uint16_t)(u >> 16);
}

#define BATCH  2
#define SEQ    2048
#define DMODEL 1024
#define NHEAD  16
#define HDIM   64
#define MTOT   (BATCH * SEQ)   // 4096

// ---------- f32 -> bf16 convert ----------
__global__ void cvt_f32_bf16_kernel(const float* __restrict__ in,
                                    uint16_t* __restrict__ out, int n) {
  int i = (blockIdx.x * blockDim.x + threadIdx.x) * 4;
  if (i + 3 < n) {
    f32x4 f = *(const f32x4*)(in + i);
    u16x4 o;
    o[0] = f2bf(f[0]); o[1] = f2bf(f[1]); o[2] = f2bf(f[2]); o[3] = f2bf(f[3]);
    *(u16x4*)(out + i) = o;
  }
}

// ---------- QKV projection GEMM: C = A @ W^T + b ----------
// A: (MTOT, DMODEL) bf16 row-major.  Wt: (DMODEL, DMODEL) bf16 (row n = K contiguous).
// transpose_out==0: out[(b*H+h)*S*W + s*W + w]   (row-major per head, for Q/K)
// transpose_out==1: out[(b*H+h)*W*S + w*S + s]   (transposed per head, for V)
__global__ __launch_bounds__(128, 1)
void qkv_gemm_kernel(const uint16_t* __restrict__ A,
                     const uint16_t* __restrict__ Wt,
                     const float*    __restrict__ bias,
                     uint16_t* __restrict__ out, int transpose_out) {
  const int lane   = threadIdx.x & 31;
  const int wid    = threadIdx.x >> 5;
  const int waveId = blockIdx.x * 4 + wid;   // 4096 wave-tiles
  const int tileM  = waveId >> 4;            // 256 tiles of 16 rows
  const int tileN  = waveId & 15;            // 16 tiles of 64 cols
  const int m0     = tileM * 16;
  const int n0     = tileN * 64;
  const int laneLo = lane & 15;
  const int laneHi = lane >> 4;

  floatx8 acc[4] = {};

  const uint16_t* arow = A + (size_t)(m0 + laneLo) * DMODEL;
  const uint16_t* wrow[4];
#pragma unroll
  for (int t = 0; t < 4; ++t)
    wrow[t] = Wt + (size_t)(n0 + t * 16 + laneLo) * DMODEL + laneHi * 8;

  auto loadA = [&](int k, BfVec& a) {
    const uint16_t* ap = arow + k + laneHi * 8;
    a.q[0] = *(const u32x4*)(ap);
    a.q[1] = *(const u32x4*)(ap + 16);
  };
  auto loadB = [&](int k, BfVec (&b)[4]) {
#pragma unroll
    for (int t = 0; t < 4; ++t) {
      b[t].q[0] = *(const u32x4*)(wrow[t] + k);
      b[t].q[1] = *(const u32x4*)(wrow[t] + k + 16);
    }
  };

  // software-pipelined: loads for step i+1 issued before WMMAs of step i
  BfVec a0, a1;
  BfVec b0[4], b1[4];
  loadA(0, a0);
  loadB(0, b0);
  for (int k = 0; k < DMODEL; k += 64) {
    loadA(k + 32, a1);
    loadB(k + 32, b1);
#pragma unroll
    for (int t = 0; t < 4; ++t)
      acc[t] = __builtin_amdgcn_wmma_f32_16x16x32_bf16(
          false, a0.v, false, b0[t].v, (short)0, acc[t], false, false);
    const int k2 = (k + 64 < DMODEL) ? (k + 64) : 0;   // wrap: avoids OOB, result unused
    loadA(k2, a0);
    loadB(k2, b0);
#pragma unroll
    for (int t = 0; t < 4; ++t)
      acc[t] = __builtin_amdgcn_wmma_f32_16x16x32_bf16(
          false, a1.v, false, b1[t].v, (short)0, acc[t], false, false);
  }

  const int batch = m0 >> 11;                 // m0 / 2048
  const int sBase = (m0 & 2047) + laneHi * 8; // C-layout: VGPR r -> row r (+8 hi half)
#pragma unroll
  for (int t = 0; t < 4; ++t) {
    const int n = n0 + t * 16 + laneLo;
    const int h = n >> 6, w = n & 63;
    const float bb = bias[n];
    if (!transpose_out) {
      const size_t base = ((size_t)(batch * NHEAD + h) * SEQ) * HDIM + w;
#pragma unroll
      for (int r = 0; r < 8; ++r)
        out[base + (size_t)(sBase + r) * HDIM] = f2bf(acc[t][r] + bb);
    } else {
      Pack8 pk;
#pragma unroll
      for (int r = 0; r < 8; ++r) pk.s[r] = f2bf(acc[t][r] + bb);
      const size_t base = ((size_t)(batch * NHEAD + h) * HDIM + w) * SEQ + sBase;
      *(u32x4*)(out + base) = pk.q;           // 8 contiguous bf16 = one 16B store
    }
  }
}

// ---------- flash attention: one wave per (b,h, 16-row q tile) ----------
__global__ __launch_bounds__(128, 1)
void attn_kernel(const uint16_t* __restrict__ Q,   // (B*H, S, W) bf16
                 const uint16_t* __restrict__ K,   // (B*H, S, W) bf16
                 const uint16_t* __restrict__ Vt,  // (B*H, W, S) bf16
                 const int*      __restrict__ mask,// (B, S)
                 float*          __restrict__ out) // (B, S, D) f32
{
  __shared__ uint16_t pbuf[4][16][40];  // padded: row stride 80B (16B aligned, conflict-free)
  const int lane   = threadIdx.x & 31;
  const int wid    = threadIdx.x >> 5;
  const int waveId = blockIdx.x * 4 + wid;   // 4096 = 32 bh * 128 q-tiles
  const int bh     = waveId >> 7;
  const int qt     = waveId & 127;
  const int batch  = bh >> 4;
  const int h      = bh & 15;
  const int s0     = qt * 16;
  const int laneLo = lane & 15;
  const int laneHi = lane >> 4;

  const uint16_t* qh   = Q  + (size_t)bh * SEQ * HDIM;
  const uint16_t* kh   = K  + (size_t)bh * SEQ * HDIM;
  const uint16_t* vh   = Vt + (size_t)bh * HDIM * SEQ;
  const int*      mrow = mask + batch * SEQ;

  // preload q tile in A-operand layout (two K-chunks of 32 covering W=64)
  BfVec aq[2];
  {
    const uint16_t* qr = qh + (size_t)(s0 + laneLo) * HDIM;
#pragma unroll
    for (int c = 0; c < 2; ++c) {
      const uint16_t* p = qr + c * 32 + laneHi * 8;
      aq[c].q[0] = *(const u32x4*)(p);
      aq[c].q[1] = *(const u32x4*)(p + 16);
    }
  }

  floatx8 acc[4] = {};
  float m_run[8], l_run[8];
#pragma unroll
  for (int r = 0; r < 8; ++r) { m_run[r] = -3.0e38f; l_run[r] = 0.0f; }

  auto loadK16 = [&](int kv, BfVec (&bk)[2][2]) {
#pragma unroll
    for (int j = 0; j < 2; ++j) {
      const uint16_t* kr = kh + (size_t)(kv + j * 16 + laneLo) * HDIM + laneHi * 8;
#pragma unroll
      for (int c = 0; c < 2; ++c) {
        const uint16_t* p = kr + c * 32;
        bk[j][c].q[0] = *(const u32x4*)(p);
        bk[j][c].q[1] = *(const u32x4*)(p + 16);
      }
    }
  };

  auto chunk = [&](int kv0, BfVec (&cur)[2][2], BfVec (&nxt)[2][2]) {
    // V-operand loads first: independent of scores, overlap softmax VALU
    BfVec bv[4];
#pragma unroll
    for (int t = 0; t < 4; ++t) {
      const uint16_t* vr = vh + (size_t)(t * 16 + laneLo) * SEQ + kv0 + laneHi * 8;
      bv[t].q[0] = *(const u32x4*)(vr);
      bv[t].q[1] = *(const u32x4*)(vr + 16);
    }

    // scores: 2 tiles of 16 kv columns, K-dim = W = 64 -> 2 WMMAs each
    floatx8 sc[2] = {};
#pragma unroll
    for (int j = 0; j < 2; ++j)
#pragma unroll
      for (int c = 0; c < 2; ++c)
        sc[j] = __builtin_amdgcn_wmma_f32_16x16x32_bf16(
            false, aq[c].v, false, cur[j][c].v, (short)0, sc[j], false, false);

    // issue next chunk's K loads now: hidden behind softmax + LDS + PV WMMAs
    int kn = kv0 + 32;
    if (kn >= SEQ) kn = 0;                   // wrap: valid address, result unused
    loadK16(kn, nxt);

    const float pen0 = mrow[kv0 + laneLo]      ? 0.0f : -10000.0f;
    const float pen1 = mrow[kv0 + 16 + laneLo] ? 0.0f : -10000.0f;

    float f0[8], f1[8], corr[8];
#pragma unroll
    for (int r = 0; r < 8; ++r) {
      f0[r] = sc[0][r] * 0.125f + pen0;      // 1/sqrt(64) = 0.125
      f1[r] = sc[1][r] * 0.125f + pen1;
    }

    // online softmax: row reductions across the 16 lanes of each half-wave
#pragma unroll
    for (int r = 0; r < 8; ++r) {
      float vmax = fmaxf(f0[r], f1[r]);
      vmax = fmaxf(vmax, __shfl_xor(vmax, 8, 16));
      vmax = fmaxf(vmax, __shfl_xor(vmax, 4, 16));
      vmax = fmaxf(vmax, __shfl_xor(vmax, 2, 16));
      vmax = fmaxf(vmax, __shfl_xor(vmax, 1, 16));
      const float mo = m_run[r];
      const float mn = fmaxf(mo, vmax);
      const float c0 = __expf(mo - mn);
      const float p0 = __expf(f0[r] - mn);
      const float p1 = __expf(f1[r] - mn);
      float rs = p0 + p1;
      rs += __shfl_xor(rs, 8, 16);
      rs += __shfl_xor(rs, 4, 16);
      rs += __shfl_xor(rs, 2, 16);
      rs += __shfl_xor(rs, 1, 16);
      l_run[r] = l_run[r] * c0 + rs;
      m_run[r] = mn;
      corr[r]  = c0;
      const int prow = r + laneHi * 8;       // C-layout row for this VGPR slot
      pbuf[wid][prow][laneLo]      = f2bf(p0);
      pbuf[wid][prow][16 + laneLo] = f2bf(p1);
    }

#pragma unroll
    for (int t = 0; t < 4; ++t)
#pragma unroll
      for (int r = 0; r < 8; ++r) acc[t][r] *= corr[r];

    // reload P from LDS in A-operand layout (DS ops are in-order within a wave)
    BfVec ap;
    {
      const uint16_t* p = &pbuf[wid][laneLo][laneHi * 8];
      ap.q[0] = *(const u32x4*)(p);
      ap.q[1] = *(const u32x4*)(p + 16);
    }

    // P @ V chunk: V^T layout makes B-operand columns contiguous
#pragma unroll
    for (int t = 0; t < 4; ++t)
      acc[t] = __builtin_amdgcn_wmma_f32_16x16x32_bf16(
          false, ap.v, false, bv[t].v, (short)0, acc[t], false, false);
  };

  // ping-pong K operand buffers across 32-wide KV chunks
  BfVec bkA[2][2], bkB[2][2];
  loadK16(0, bkA);
  for (int kv0 = 0; kv0 < SEQ; kv0 += 64) {
    chunk(kv0,      bkA, bkB);
    chunk(kv0 + 32, bkB, bkA);
  }

  // epilogue: out[b][s][h*64 + w] = acc / l
  float* ob = out + (size_t)batch * SEQ * DMODEL + (size_t)h * HDIM;
#pragma unroll
  for (int t = 0; t < 4; ++t) {
    const int w = t * 16 + laneLo;
#pragma unroll
    for (int r = 0; r < 8; ++r) {
      const int s = s0 + r + laneHi * 8;
      ob[(size_t)s * DMODEL + w] = acc[t][r] / l_run[r];
    }
  }
}

// ---------- host launcher ----------
extern "C" void kernel_launch(void* const* d_in, const int* in_sizes, int n_in,
                              void* d_out, int out_size, void* d_ws, size_t ws_size,
                              hipStream_t stream) {
  (void)in_sizes; (void)n_in; (void)out_size; (void)ws_size;
  const float* x_q  = (const float*)d_in[0];
  const float* x_k  = (const float*)d_in[1];
  const float* x_v  = (const float*)d_in[2];
  const int*   mask = (const int*)  d_in[3];
  const float* Wq   = (const float*)d_in[4];
  const float* bq   = (const float*)d_in[5];
  const float* Wk   = (const float*)d_in[6];
  const float* bk   = (const float*)d_in[7];
  const float* Wv   = (const float*)d_in[8];
  const float* bv   = (const float*)d_in[9];
  float* out = (float*)d_out;

  const size_t NX = (size_t)MTOT * DMODEL;     // 4194304
  const size_t NW = (size_t)DMODEL * DMODEL;   // 1048576

  uint8_t* ws  = (uint8_t*)d_ws;
  size_t   off = 0;
  auto take = [&](size_t bytes) -> uint16_t* {
    uint16_t* p = (uint16_t*)(ws + off);
    off += (bytes + 255) & ~(size_t)255;
    return p;
  };
  uint16_t* xq_bf = take(NX * 2);
  uint16_t* xk_bf = take(NX * 2);
  uint16_t* xv_bf = take(NX * 2);
  uint16_t* wq_bf = take(NW * 2);
  uint16_t* wk_bf = take(NW * 2);
  uint16_t* wv_bf = take(NW * 2);
  uint16_t* q_bf  = take(NX * 2);
  uint16_t* k_bf  = take(NX * 2);
  uint16_t* vT_bf = take(NX * 2);

  // 1) f32 -> bf16 conversions
  cvt_f32_bf16_kernel<<<(int)(NX / 1024), 256, 0, stream>>>(x_q, xq_bf, (int)NX);
  cvt_f32_bf16_kernel<<<(int)(NX / 1024), 256, 0, stream>>>(x_k, xk_bf, (int)NX);
  cvt_f32_bf16_kernel<<<(int)(NX / 1024), 256, 0, stream>>>(x_v, xv_bf, (int)NX);
  cvt_f32_bf16_kernel<<<(int)(NW / 1024), 256, 0, stream>>>(Wq, wq_bf, (int)NW);
  cvt_f32_bf16_kernel<<<(int)(NW / 1024), 256, 0, stream>>>(Wk, wk_bf, (int)NW);
  cvt_f32_bf16_kernel<<<(int)(NW / 1024), 256, 0, stream>>>(Wv, wv_bf, (int)NW);

  // 2) QKV projections (4096 wave-tiles each; 128 threads = 4 waves per block)
  qkv_gemm_kernel<<<1024, 128, 0, stream>>>(xq_bf, wq_bf, bq, q_bf, 0);
  qkv_gemm_kernel<<<1024, 128, 0, stream>>>(xk_bf, wk_bf, bk, k_bf, 0);
  qkv_gemm_kernel<<<1024, 128, 0, stream>>>(xv_bf, wv_bf, bv, vT_bf, 1);

  // 3) flash attention
  attn_kernel<<<1024, 128, 0, stream>>>(q_bf, k_bf, vT_bf, mask, out);
}
